// IterativeGamePool_59596966199333
// MI455X (gfx1250) — matrix operand order repaired and checked by reading, not compiled, not run on hardware
//
#include <hip/hip_runtime.h>
#include <hip/hip_bf16.h>
#include <stdint.h>

// Problem constants (from reference)
#define COM   128
#define INP   512
#define HID   2048
#define OUTD  512
#define DIN   640     // INP + COM
#define DOUT  640     // OUTD + COM
#define TSTEPS 16384
#define GSCALE 35.0f

#define NWG      80   // persistent workgroups: 40 A1-W2 tiles + 40 A2-W2 tiles
#define NTHREADS 256  // 8 waves (wave32)

typedef __bf16 bf16;
typedef __attribute__((ext_vector_type(16))) __bf16 v16bf;
typedef __attribute__((ext_vector_type(8)))  float  v8f;

// ---------------------------------------------------------------- workspace
struct WS {
  // bf16 vectors first (16B-aligned offsets for async b128 copies / v16bf loads)
  bf16 m2a[COM];  bf16 m1[COM];  bf16 m2b[COM];
  bf16 h1a[HID];  bf16 h2[HID];  bf16 h1b[HID];
  float base1[HID];  float base2[HID];
  float sum_c1[COM]; float sum_c2[COM];
  float c1[COM];     float c2[COM];
  unsigned bar_cnt;  unsigned bar_gen;
};

// A-fragment packing for V_WMMA_F32_16X16X32_BF16 (16-bit A 16x32, contiguous k = 16*(lane/16) + i, same order as B):
// packed index p = chunk*512 + lane*16 + i  ->  (row M, column k)
__device__ __host__ inline void frag_a_coords(int p, int& M, int& k) {
  int chunk = p >> 9;
  int lane  = (p >> 4) & 31;
  int i     = p & 15;
  int kk;
  if (lane < 16) { M = lane;      kk = i;  }
  else           { M = lane - 16; kk = i + 16; }
  k = chunk * 32 + kk;
}

__device__ inline float lrelu(float v) { return v > 0.f ? v : 0.01f * v; }

// ---------------------------------------------------------------- prep
__global__ void prep_base(const float* __restrict__ W1, const float* __restrict__ b1,
                          const float* __restrict__ I, float* __restrict__ baseOut) {
  int r = blockIdx.x * blockDim.x + threadIdx.x;
  if (r >= HID) return;
  const float* wr = W1 + (size_t)r * DIN;
  float acc = b1[r];
  for (int k = 0; k < INP; ++k) acc += wr[k] * I[k];
  baseOut[r] = acc;
}

__global__ void init_ws(WS* ws) {
  int i = threadIdx.x;
  if (i < COM) {
    ws->m2a[i] = (bf16)0.f; ws->m1[i] = (bf16)0.f; ws->m2b[i] = (bf16)0.f;
    ws->sum_c1[i] = 0.f; ws->sum_c2[i] = 0.f;
    ws->c1[i] = 0.f;     ws->c2[i] = 0.f;
  }
  if (i == 0) { ws->bar_cnt = 0u; ws->bar_gen = 0u; }
}

// ---------------------------------------------------------------- persistent kernel
__global__ void __launch_bounds__(NTHREADS, 1)
game_kernel(const float* __restrict__ A1W1, const float* __restrict__ A1W2,
            const float* __restrict__ A1b2,
            const float* __restrict__ A2W1, const float* __restrict__ A2W2,
            const float* __restrict__ A2b2,
            WS* ws, float* __restrict__ out) {
  const int w    = blockIdx.x;
  const int lane = threadIdx.x & 31;
  const int wv   = threadIdx.x >> 5;

  // LDS-resident weight tiles, packed in WMMA A-fragment order.
  __shared__ __align__(16) bf16  lds_w2[16 * HID];      // 64 KB: 16-row x 2048-K W2 tile
  __shared__ __align__(16) bf16  lds_w1m[4 * 16 * COM]; // 16 KB: up to 2 L1 tiles/agent
  __shared__ __align__(16) bf16  lds_vec[HID];          //  4 KB: broadcast m/h vector
  __shared__ float lds_red[8 * 16];                     // cross-wave K-reduction
  __shared__ float lds_sm[COM];                         // softmax scratch (WG 0)

  // ---- stage layer-1 slices (message part of W1: columns [INP, DIN))
  const int nt1 = (w < 48) ? 2 : 1;         // 128 tiles of 16 rows over 80 WGs
  for (int a = 0; a < 2; ++a) {
    const float* W1 = a ? A2W1 : A1W1;
    for (int s = 0; s < nt1; ++s) {
      const int tile = (s == 0) ? w : (w + NWG);
      bf16* dst = &lds_w1m[(a * 2 + s) * (16 * COM)];
      for (int p = threadIdx.x; p < 16 * COM; p += NTHREADS) {
        int M, k; frag_a_coords(p, M, k);
        dst[p] = (bf16)W1[(size_t)(tile * 16 + M) * DIN + INP + k];
      }
    }
  }
  // ---- stage this WG's W2 tile (WGs 0..39 -> agent1 rows, 40..79 -> agent2 rows)
  const float* W2g    = (w < 40) ? A1W2 : A2W2;
  const int rowBase2  = ((w < 40) ? w : (w - 40)) * 16;
  for (int p = threadIdx.x; p < 16 * HID; p += NTHREADS) {
    int M, k; frag_a_coords(p, M, k);
    lds_w2[p] = (bf16)W2g[(size_t)(rowBase2 + M) * HID + k];
  }
  __syncthreads();

  unsigned* cnt = &ws->bar_cnt;
  unsigned* gen = &ws->bar_gen;

  auto grid_barrier = [&]() {
    __threadfence();
    __syncthreads();
    if (threadIdx.x == 0) {
      unsigned g = __hip_atomic_load(gen, __ATOMIC_ACQUIRE, __HIP_MEMORY_SCOPE_AGENT);
      unsigned a = __hip_atomic_fetch_add(cnt, 1u, __ATOMIC_ACQ_REL, __HIP_MEMORY_SCOPE_AGENT);
      if (a == NWG - 1) {
        __hip_atomic_store(cnt, 0u, __ATOMIC_RELAXED, __HIP_MEMORY_SCOPE_AGENT);
        __hip_atomic_fetch_add(gen, 1u, __ATOMIC_RELEASE, __HIP_MEMORY_SCOPE_AGENT);
      } else {
        while (__hip_atomic_load(gen, __ATOMIC_ACQUIRE, __HIP_MEMORY_SCOPE_AGENT) == g)
          __builtin_amdgcn_s_sleep(1);
      }
    }
    __syncthreads();
  };

  // Broadcast a global bf16 vector into lds_vec via async DMA (ASYNCcnt-tracked,
  // no VGPR staging): wave 0 issues one b128 transfer per lane per 512B.
  auto stage_vec = [&](const bf16* g, int nbytes) {
    if (wv == 0) {
      const int nchunks = nbytes >> 4;     // 16B units
      const uint32_t lds_base = (uint32_t)(uintptr_t)(&lds_vec[0]);
      const uint64_t g_base   = (uint64_t)(uintptr_t)g;
      for (int c = lane; c < nchunks; c += 32) {
        uint32_t ldsoff = lds_base + (uint32_t)(c << 4);
        uint64_t gaddr  = g_base + (uint64_t)(c << 4);
        asm volatile("global_load_async_to_lds_b128 %0, %1, off"
                     :: "v"(ldsoff), "v"(gaddr) : "memory");
      }
      asm volatile("s_wait_asynccnt 0x0" ::: "memory");
    }
    __syncthreads();
  };

  // h[2048] = lrelu(base + W1m @ m); vector broadcast in all 16 B columns, D col 0 used.
  // Two independent accumulator chains to halve the dependent-WMMA critical path.
  auto layer1 = [&](int a, const float* base_g, bf16* h) {
    if (wv < nt1) {
      const int tile = (wv == 0) ? w : (w + NWG);
      const bf16* lw = &lds_w1m[(a * 2 + wv) * (16 * COM)];
      const int boff = (lane & 16) ? 16 : 0;
      v8f acc0 = {}, acc1 = {};
      #pragma unroll
      for (int c = 0; c < COM / 32; c += 2) {
        v16bf a0 = *(const v16bf*)(lw + (c + 0) * 512 + lane * 16);
        v16bf b0 = *(const v16bf*)(lds_vec + (c + 0) * 32 + boff);
        acc0 = __builtin_amdgcn_wmma_f32_16x16x32_bf16(false, a0, false, b0,
                                                       (short)0, acc0, false, false);
        v16bf a1 = *(const v16bf*)(lw + (c + 1) * 512 + lane * 16);
        v16bf b1 = *(const v16bf*)(lds_vec + (c + 1) * 32 + boff);
        acc1 = __builtin_amdgcn_wmma_f32_16x16x32_bf16(false, a1, false, b1,
                                                       (short)0, acc1, false, false);
      }
      if ((lane & 15) == 0) {
        const int rb = tile * 16 + (lane >> 4) * 8;
        #pragma unroll
        for (int j = 0; j < 8; ++j)
          h[rb + j] = (bf16)lrelu(base_g[rb + j] + acc0[j] + acc1[j]);
      }
    }
  };

  // 16-row slice of W2 @ h; K=2048 over 8 waves x (2 chains x 4 WMMA), LDS reduce.
  auto layer2 = [&]() {
    const int boff = (lane & 16) ? 16 : 0;
    v8f acc0 = {}, acc1 = {};
    for (int c = wv; c < HID / 32; c += 16) {
      v16bf a0 = *(const v16bf*)(lds_w2 + c * 512 + lane * 16);
      v16bf b0 = *(const v16bf*)(lds_vec + c * 32 + boff);
      acc0 = __builtin_amdgcn_wmma_f32_16x16x32_bf16(false, a0, false, b0,
                                                     (short)0, acc0, false, false);
      v16bf a1 = *(const v16bf*)(lds_w2 + (c + 8) * 512 + lane * 16);
      v16bf b1 = *(const v16bf*)(lds_vec + (c + 8) * 32 + boff);
      acc1 = __builtin_amdgcn_wmma_f32_16x16x32_bf16(false, a1, false, b1,
                                                     (short)0, acc1, false, false);
    }
    if ((lane & 15) == 0) {
      const int off = (lane >> 4) * 8;
      #pragma unroll
      for (int j = 0; j < 8; ++j) lds_red[wv * 16 + off + j] = acc0[j] + acc1[j];
    }
    __syncthreads();
  };

  auto layer2_row = [&](const float* b2v) -> float { // valid for threads < 16
    float val = 0.f;
    if (threadIdx.x < 16) {
      float s = 0.f;
      #pragma unroll
      for (int v = 0; v < 8; ++v) s += lds_red[v * 16 + threadIdx.x];
      val = lrelu(s + b2v[rowBase2 + threadIdx.x]);
    }
    return val;
  };

  // WG0: com = softmax(SCALE*logits); sum_c += com; m = sum_c/denom (bf16)
  auto softmax_update = [&](const float* logits, float* sum_c, float dcur, float dnext,
                            bf16* mcur, bf16* mnext) {
    const int i = threadIdx.x;
    float v = -1e30f;
    if (i < COM) { v = GSCALE * logits[i]; lds_sm[i] = v; }
    __syncthreads();
    for (int s = COM / 2; s > 0; s >>= 1) {
      if (i < s) lds_sm[i] = fmaxf(lds_sm[i], lds_sm[i + s]);
      __syncthreads();
    }
    const float mx = lds_sm[0];
    __syncthreads();
    float e = 0.f;
    if (i < COM) { e = __expf(v - mx); lds_sm[i] = e; }
    __syncthreads();
    for (int s = COM / 2; s > 0; s >>= 1) {
      if (i < s) lds_sm[i] += lds_sm[i + s];
      __syncthreads();
    }
    const float tot = lds_sm[0];
    if (i < COM) {
      float sc = sum_c[i] + e / tot;
      sum_c[i] = sc;
      mcur[i] = (bf16)(sc / dcur);
      if (mnext) mnext[i] = (bf16)(sc / dnext);
    }
    __syncthreads();
  };

  float* out1 = out;
  float* out2 = out + (size_t)TSTEPS * OUTD;

  grid_barrier();  // staging + init visible everywhere

  for (int t = 0; t < TSTEPS; ++t) {
    const float denom = (float)(t + 1);

    // S1: h1a = lrelu(base1 + W1m1 @ m2a)
    stage_vec(ws->m2a, COM * 2);
    layer1(0, ws->base1, ws->h1a);
    grid_barrier();

    // S2: com1 logits (agent1 W2 rows 512..639 -> WGs 32..39)
    if (w >= 32 && w < 40) {
      stage_vec(ws->h1a, HID * 2);
      layer2();
      float v = layer2_row(A1b2);
      if (threadIdx.x < 16) ws->c1[rowBase2 + (int)threadIdx.x - OUTD] = v;
    }
    grid_barrier();

    // S3: softmax -> sum_c1, m1
    if (w == 0) softmax_update(ws->c1, ws->sum_c1, denom, 0.f, ws->m1, nullptr);
    grid_barrier();

    // S4: h2 = lrelu(base2 + W1m2 @ m1)
    stage_vec(ws->m1, COM * 2);
    layer1(1, ws->base2, ws->h2);
    grid_barrier();

    // S5: full2 (agent2 W2, all 640 rows -> WGs 40..79); out2 stored directly
    if (w >= 40) {
      stage_vec(ws->h2, HID * 2);
      layer2();
      float v = layer2_row(A2b2);
      if (threadIdx.x < 16) {
        const int row = rowBase2 + (int)threadIdx.x;
        if (row < OUTD) out2[(size_t)t * OUTD + row] = v;
        else            ws->c2[row - OUTD] = v;
      }
    }
    grid_barrier();

    // S6: softmax -> sum_c2, m2b (this step) and m2a (next step, denom+1)
    if (w == 0) softmax_update(ws->c2, ws->sum_c2, denom, denom + 1.f, ws->m2b, ws->m2a);
    grid_barrier();

    // S7: h1b = lrelu(base1 + W1m1 @ m2b)
    stage_vec(ws->m2b, COM * 2);
    layer1(0, ws->base1, ws->h1b);
    grid_barrier();

    // S8: out1 (agent1 W2 rows 0..511 -> WGs 0..31); no trailing barrier needed:
    // next conflicting write (h1b in S7 of t+1) is separated by >=6 barriers.
    if (w < 32) {
      stage_vec(ws->h1b, HID * 2);
      layer2();
      float v = layer2_row(A1b2);
      if (threadIdx.x < 16) out1[(size_t)t * OUTD + rowBase2 + threadIdx.x] = v;
    }
  }
}

// ---------------------------------------------------------------- launch
extern "C" void kernel_launch(void* const* d_in, const int* in_sizes, int n_in,
                              void* d_out, int out_size, void* d_ws, size_t ws_size,
                              hipStream_t stream) {
  (void)in_sizes; (void)n_in; (void)out_size; (void)ws_size;
  const float* I1     = (const float*)d_in[0];
  const float* I2     = (const float*)d_in[1];
  const float* A1W1_w = (const float*)d_in[2];
  const float* A1W1_b = (const float*)d_in[3];
  const float* A1W2_w = (const float*)d_in[4];
  const float* A1W2_b = (const float*)d_in[5];
  const float* A2W1_w = (const float*)d_in[6];
  const float* A2W1_b = (const float*)d_in[7];
  const float* A2W2_w = (const float*)d_in[8];
  const float* A2W2_b = (const float*)d_in[9];

  WS*    ws  = (WS*)d_ws;
  float* out = (float*)d_out;

  init_ws<<<1, 128, 0, stream>>>(ws);
  prep_base<<<(HID + 255) / 256, 256, 0, stream>>>(A1W1_w, A1W1_b, I1, ws->base1);
  prep_base<<<(HID + 255) / 256, 256, 0, stream>>>(A2W1_w, A2W1_b, I2, ws->base2);
  game_kernel<<<NWG, NTHREADS, 0, stream>>>(A1W1_w, A1W2_w, A1W2_b,
                                            A2W1_w, A2W2_w, A2W2_b, ws, out);
}
